// GATModel_455266533899
// MI455X (gfx1250) — matrix operand order repaired and checked by reading, not compiled, not run on hardware
//
#include <hip/hip_runtime.h>
#include <hip/hip_bf16.h>
#include <stdint.h>

// Problem constants (match reference)
#define NNODES 50000
#define NEDGES 800000
#define NGRAPH 50
#define NHEAD  4
#define DDIM   64
#define FHD    256      // NHEAD * DDIM
#define K0PAD  32       // 26 node features padded to 32 for WMMA K
#define DH1    512
#define NEG_SLOPE 0.2f

typedef __bf16          v16bf  __attribute__((ext_vector_type(16)));
typedef unsigned short  v16us  __attribute__((ext_vector_type(16)));
typedef float           v8f    __attribute__((ext_vector_type(8)));

// ---------- helpers ----------
static __device__ __forceinline__ unsigned short f2bf(float f) {
  unsigned u = __float_as_uint(f);
  unsigned r = u + 0x7FFFu + ((u >> 16) & 1u);   // round-to-nearest-even
  return (unsigned short)(r >> 16);
}
// monotone float <-> uint map for atomic segment-max
static __device__ __forceinline__ unsigned fenc(float f) {
  unsigned u = __float_as_uint(f);
  return (u & 0x80000000u) ? ~u : (u | 0x80000000u);
}
static __device__ __forceinline__ float fdec(unsigned k) {
  return (k & 0x80000000u) ? __uint_as_float(k & 0x7FFFFFFFu)
                           : __uint_as_float(~k);
}

// ---------- packing into WMMA fragment order (ISA 7.12.2, 16-bit 16x32 A / 32x16 B) ----------
// Packed A: per (16-row, 32-K) block: 32 lanes x 16 bf16 contiguous per lane.
// lane<16: row=lane,    K in {0..7, 16..23}   (idx<8 -> kb+idx ; idx>=8 -> kb+idx+8, kb=0)
// lane>=16: row=lane-16, K in {8..15, 24..31} (kb=8)
__global__ void packA_kernel(const float* __restrict__ in, unsigned short* __restrict__ out,
                             int M, int K, int relu) {
  int i = blockIdx.x * blockDim.x + threadIdx.x;          // packed index in bf16 units
  if (i >= M * K) return;
  int blk  = i >> 10;                                     // 1024 elems per 16x32 block
  int lane = (i & 1023) >> 4;
  int idx  = i & 15;
  int kblocks = K >> 5;
  int n   = (blk / kblocks) * 16 + (lane & 15);
  int kb  = (lane < 16) ? 0 : 8;
  int k   = (blk % kblocks) * 32 + kb + idx + ((idx >= 8) ? 8 : 0);
  float v = in[(size_t)n * K + k];
  if (relu) v = fmaxf(v, 0.f);
  out[i] = f2bf(v);
}

// Packed B: per (32-K, 16-col) block: lane<16 -> n=lane, K in {0..7, 16..23} ; lane>=16 -> n=lane-16, K in {8..15, 24..31}
__global__ void packB_kernel(const float* __restrict__ W, unsigned short* __restrict__ out,
                             int K, int P, int Korig) {
  int i = blockIdx.x * blockDim.x + threadIdx.x;
  if (i >= K * P) return;
  int blk  = i >> 10;
  int lane = (i & 1023) >> 4;
  int idx  = i & 15;
  int pblocks = P >> 4;
  int n = (blk % pblocks) * 16 + (lane & 15);
  int k = (blk / pblocks) * 32 + ((lane < 16) ? 0 : 8) + idx + ((idx >= 8) ? 8 : 0);
  float v = (k < Korig) ? W[(size_t)k * P + n] : 0.f;     // zero-pad W0 rows 26..31
  out[i] = f2bf(v);
}

// ---------- WMMA GEMM: C[MxP] = A(bf16 packed) x B(bf16 packed), f32 accum ----------
__global__ void __launch_bounds__(256)
gemm_bf16_wmma(const unsigned short* __restrict__ A, const unsigned short* __restrict__ B,
               float* __restrict__ C, int M, int K, int P,
               const float* __restrict__ bias, int relu) {
  const int lane = threadIdx.x & 31;
  const int wave = threadIdx.x >> 5;
  const int tilesP = P >> 4;
  const int tile = blockIdx.x * 8 + wave;
  if (tile >= (M >> 4) * tilesP) return;                  // wave-uniform exit (EXEC stays full)
  const int m0 = (tile / tilesP) << 4;
  const int n0 = (tile % tilesP) << 4;
  const int kblocks = K >> 5;
  const unsigned short* Ap = A + ((size_t)(m0 >> 4) * kblocks) * 1024 + lane * 16;
  const unsigned short* Bp = B + ((size_t)(n0 >> 4)) * 1024 + lane * 16;
  const size_t bstride = (size_t)tilesP * 1024;
  v8f acc = {0.f, 0.f, 0.f, 0.f, 0.f, 0.f, 0.f, 0.f};
  for (int kb = 0; kb < kblocks; ++kb) {
    v16us au = *(const v16us*)(Ap + (size_t)kb * 1024);   // 32B contiguous per lane
    v16us bu = *(const v16us*)(Bp + (size_t)kb * bstride);
    v16bf a = __builtin_bit_cast(v16bf, au);
    v16bf b = __builtin_bit_cast(v16bf, bu);
    acc = __builtin_amdgcn_wmma_f32_16x16x32_bf16(false, a, false, b, (short)0, acc,
                                                  false, false);
  }
  const int col = n0 + (lane & 15);
  float bv = bias ? bias[col] : 0.f;
#pragma unroll
  for (int v = 0; v < 8; ++v) {
    int row = m0 + v + ((lane < 16) ? 0 : 8);             // ISA C/D layout
    float val = acc[v] + bv;
    if (relu) val = fmaxf(val, 0.f);
    C[(size_t)row * P + col] = val;
  }
}

// ---------- feature assembly / small kernels ----------
__global__ void embed_kernel(const int* __restrict__ seq, const float* __restrict__ node_s,
                             const float* __restrict__ W_emb, float* __restrict__ h0) {
  int i = blockIdx.x * blockDim.x + threadIdx.x;          // i = n*32 + j
  if (i >= NNODES * K0PAD) return;
  int n = i >> 5, j = i & 31;
  float v = 0.f;
  if (j < 20)      v = W_emb[seq[n] * 20 + j];
  else if (j < 26) v = node_s[n * 6 + (j - 20)];
  h0[i] = v;
}

__global__ void scores_kernel(const float* __restrict__ ft, const float* __restrict__ al,
                              const float* __restrict__ ar, float* __restrict__ el,
                              float* __restrict__ er) {
  int i = blockIdx.x * blockDim.x + threadIdx.x;          // i = n*4 + h
  if (i >= NNODES * NHEAD) return;
  int h = i & 3, n = i >> 2;
  const float* f  = ft + (size_t)n * FHD + h * DDIM;
  const float* a1 = al + h * DDIM;
  const float* a2 = ar + h * DDIM;
  float s1 = 0.f, s2 = 0.f;
  for (int d = 0; d < DDIM; ++d) { float fv = f[d]; s1 += fv * a1[d]; s2 += fv * a2[d]; }
  el[i] = s1; er[i] = s2;
}

__global__ void init_bias_kernel(float* __restrict__ out, const float* __restrict__ b) {
  int i = blockIdx.x * blockDim.x + threadIdx.x;
  if (i < NNODES * FHD) out[i] = b[i & (FHD - 1)];
}

__global__ void zero_u32_kernel(unsigned* __restrict__ p, int n) {
  int i = blockIdx.x * blockDim.x + threadIdx.x;
  if (i < n) p[i] = 0u;
}

// edge pass 1: e = leaky_relu(el[src]+er[dst]); segment max via encoded atomicMax
__global__ void edge_max_kernel(const int* __restrict__ src, const int* __restrict__ dst,
                                const float* __restrict__ el, const float* __restrict__ er,
                                float* __restrict__ ebuf, unsigned* __restrict__ mEnc) {
  int i = blockIdx.x * blockDim.x + threadIdx.x;          // i = e*4 + h
  if (i >= NEDGES * NHEAD) return;
  int h = i & 3, e = i >> 2;
  float x = el[src[e] * 4 + h] + er[dst[e] * 4 + h];
  x = (x > 0.f) ? x : NEG_SLOPE * x;
  ebuf[i] = x;
  atomicMax(&mEnc[dst[e] * 4 + h], fenc(x));
}

// edge pass 2: ee = exp(e - m[dst]); z[dst] += ee
__global__ void edge_exp_kernel(const int* __restrict__ dst, float* __restrict__ ebuf,
                                const unsigned* __restrict__ mEnc, float* __restrict__ z) {
  int i = blockIdx.x * blockDim.x + threadIdx.x;
  if (i >= NEDGES * NHEAD) return;
  int h = i & 3, d = dst[i >> 2];
  float ee = __expf(ebuf[i] - fdec(mEnc[d * 4 + h]));
  ebuf[i] = ee;
  atomicAdd(&z[d * 4 + h], ee);
}

// edge pass 3: out[dst] += ft[src] * alpha   (one block of 256 threads per edge)
__global__ void __launch_bounds__(256)
edge_msg_kernel(const int* __restrict__ src, const int* __restrict__ dst,
                const float* __restrict__ ft, const float* __restrict__ ebuf,
                const float* __restrict__ z, float* __restrict__ out) {
  int e = blockIdx.x;
  int c = threadIdx.x;
  int s = src[e], d = dst[e];
  int h = c >> 6;
  float alpha = ebuf[e * 4 + h] / z[d * 4 + h];
  atomicAdd(&out[(size_t)d * FHD + c], ft[(size_t)s * FHD + c] * alpha);
}

// decoder tail: out[n] = hidden[n,:]·Wd2 + bd2 + 0.5 ; per-graph sums
__global__ void __launch_bounds__(256)
final_kernel(const float* __restrict__ hidden, const float* __restrict__ Wd2,
             const float* __restrict__ bd2, const int* __restrict__ graph_id,
             float* __restrict__ out, float* __restrict__ g_sum, float* __restrict__ g_cnt) {
  int wave = threadIdx.x >> 5, lane = threadIdx.x & 31;
  int n = blockIdx.x * 8 + wave;
  if (n >= NNODES) return;
  const float* hr = hidden + (size_t)n * DH1;
  float s = 0.f;
  for (int j = lane; j < DH1; j += 32) s += hr[j] * Wd2[j];
  for (int off = 16; off > 0; off >>= 1) s += __shfl_xor(s, off, 32);
  if (lane == 0) {
    float val = s + bd2[0] + 0.5f;
    out[n] = val;
    int g = graph_id[n];
    atomicAdd(&g_sum[g], val);
    atomicAdd(&g_cnt[g], 1.0f);
  }
}

__global__ void gout_kernel(const float* __restrict__ g_sum, const float* __restrict__ g_cnt,
                            float* __restrict__ gout) {
  int g = threadIdx.x;
  if (g < NGRAPH) gout[g] = g_sum[g] / g_cnt[g];
}

// ---------- host orchestration ----------
static inline unsigned cdiv(long a, long b) { return (unsigned)((a + b - 1) / b); }

extern "C" void kernel_launch(void* const* d_in, const int* in_sizes, int n_in,
                              void* d_out, int out_size, void* d_ws, size_t ws_size,
                              hipStream_t stream) {
  (void)in_sizes; (void)n_in; (void)out_size;
  const int*   seq     = (const int*)d_in[0];
  const float* node_s  = (const float*)d_in[1];
  const int*   src     = (const int*)d_in[2];
  const int*   dst     = (const int*)d_in[3];
  const int*   graphid = (const int*)d_in[4];
  const float* W_emb   = (const float*)d_in[5];
  const float* Wl[3]   = {(const float*)d_in[6],  (const float*)d_in[10], (const float*)d_in[14]};
  const float* al[3]   = {(const float*)d_in[7],  (const float*)d_in[11], (const float*)d_in[15]};
  const float* ar[3]   = {(const float*)d_in[8],  (const float*)d_in[12], (const float*)d_in[16]};
  const float* bl[3]   = {(const float*)d_in[9],  (const float*)d_in[13], (const float*)d_in[17]};
  const float* Wd1     = (const float*)d_in[18];
  const float* bd1     = (const float*)d_in[19];
  const float* Wd2     = (const float*)d_in[20];
  const float* bd2     = (const float*)d_in[21];
  float* out  = (float*)d_out;           // [N] node outputs
  float* gout = out + NNODES;            // [G] graph outputs

  // workspace carving (~247 MB total)
  char* p = (char*)d_ws;
  auto carve = [&](size_t bytes) { char* r = p; p += (bytes + 255) & ~(size_t)255; return r; };
  float*          bufA   = (float*)carve((size_t)NNODES * FHD * 4);   // h / GAT output
  float*          bufB   = (float*)carve((size_t)NNODES * FHD * 4);   // ft
  unsigned short* hb     = (unsigned short*)carve((size_t)NNODES * FHD * 2); // packed bf16 A
  float*          hidden = (float*)carve((size_t)NNODES * DH1 * 4);
  float*          el     = (float*)carve((size_t)NNODES * NHEAD * 4);
  float*          er     = (float*)carve((size_t)NNODES * NHEAD * 4);
  unsigned*       mEnc   = (unsigned*)carve((size_t)NNODES * NHEAD * 4);
  float*          z      = (float*)carve((size_t)NNODES * NHEAD * 4); // adjacent to mEnc
  float*          ebuf   = (float*)carve((size_t)NEDGES * NHEAD * 4);
  unsigned short* Wb0    = (unsigned short*)carve((size_t)K0PAD * FHD * 2);
  unsigned short* Wb1    = (unsigned short*)carve((size_t)FHD * FHD * 2);
  unsigned short* Wb2    = (unsigned short*)carve((size_t)FHD * FHD * 2);
  unsigned short* Wbd1   = (unsigned short*)carve((size_t)FHD * DH1 * 2);
  float*          g_sum  = (float*)carve((size_t)NGRAPH * 4);
  float*          g_cnt  = (float*)carve((size_t)NGRAPH * 4);
  (void)ws_size;
  unsigned short* Wb[3] = {Wb0, Wb1, Wb2};

  const int T = 256;
  // pack weights to bf16 WMMA-fragment layout
  packB_kernel<<<cdiv((long)K0PAD * FHD, T), T, 0, stream>>>(Wl[0], Wb0, K0PAD, FHD, 26);
  packB_kernel<<<cdiv((long)FHD * FHD, T), T, 0, stream>>>(Wl[1], Wb1, FHD, FHD, FHD);
  packB_kernel<<<cdiv((long)FHD * FHD, T), T, 0, stream>>>(Wl[2], Wb2, FHD, FHD, FHD);
  packB_kernel<<<cdiv((long)FHD * DH1, T), T, 0, stream>>>(Wd1, Wbd1, FHD, DH1, FHD);

  // h0 = concat(W_emb[seq], node_s) padded to 32 (fp32 staging in bufA)
  embed_kernel<<<cdiv((long)NNODES * K0PAD, T), T, 0, stream>>>(seq, node_s, W_emb, bufA);

  for (int l = 0; l < 3; ++l) {
    const int K = (l == 0) ? K0PAD : FHD;
    packA_kernel<<<cdiv((long)NNODES * K, T), T, 0, stream>>>(bufA, hb, NNODES, K, 0);
    gemm_bf16_wmma<<<cdiv((long)(NNODES / 16) * (FHD / 16), 8), T, 0, stream>>>(
        hb, Wb[l], bufB, NNODES, K, FHD, nullptr, 0);
    scores_kernel<<<cdiv((long)NNODES * NHEAD, T), T, 0, stream>>>(bufB, al[l], ar[l], el, er);
    init_bias_kernel<<<cdiv((long)NNODES * FHD, T), T, 0, stream>>>(bufA, bl[l]);
    zero_u32_kernel<<<cdiv((long)2 * NNODES * NHEAD, T), T, 0, stream>>>(mEnc, 2 * NNODES * NHEAD);
    edge_max_kernel<<<cdiv((long)NEDGES * NHEAD, T), T, 0, stream>>>(src, dst, el, er, ebuf, mEnc);
    edge_exp_kernel<<<cdiv((long)NEDGES * NHEAD, T), T, 0, stream>>>(dst, ebuf, mEnc, z);
    edge_msg_kernel<<<NEDGES, T, 0, stream>>>(src, dst, bufB, ebuf, z, bufA);
  }

  // relu(h) -> bf16 packed, decoder GEMM (fused bias+relu), tail
  packA_kernel<<<cdiv((long)NNODES * FHD, T), T, 0, stream>>>(bufA, hb, NNODES, FHD, 1);
  gemm_bf16_wmma<<<cdiv((long)(NNODES / 16) * (DH1 / 16), 8), T, 0, stream>>>(
      hb, Wbd1, hidden, NNODES, FHD, DH1, bd1, 1);
  zero_u32_kernel<<<1, T, 0, stream>>>((unsigned*)g_sum, 2 * NGRAPH);
  final_kernel<<<cdiv(NNODES, 8), T, 0, stream>>>(hidden, Wd2, bd2, graphid, out, g_sum, g_cnt);
  gout_kernel<<<1, 64, 0, stream>>>(g_sum, g_cnt, gout);
}
